// AttentionDecoder_34316788695644
// MI455X (gfx1250) — matrix operand order repaired and checked by reading, not compiled, not run on hardware
//
#include <hip/hip_runtime.h>
#include <hip/hip_bf16.h>
#include <math.h>

typedef __attribute__((ext_vector_type(16))) __bf16 v16bf;
typedef __attribute__((ext_vector_type(8)))  __bf16 v8bf;
typedef __attribute__((ext_vector_type(8)))  float  v8f;

static constexpr int Bc = 256;
static constexpr int Lc = 49;
static constexpr int Vc = 10000;
static constexpr int VcPad = 10016;      // Vc rounded up to 32 for N-tiling
static constexpr int Ec = 512;
static constexpr int Hc = 512;
static constexpr int Kc = 512;
static constexpr int Tc = 15;

// ---------------- device helpers ----------------

__device__ __forceinline__ __bf16 f2bf(float f) {
  union { float f; unsigned u; } v; v.f = f;
  unsigned r = v.u + 0x7FFFu + ((v.u >> 16) & 1u);   // round-to-nearest-even
  unsigned short h = (unsigned short)(r >> 16);
  return __builtin_bit_cast(__bf16, h);
}

// A-fragment (16x32 bf16) per ISA layout: lanes 0-15 -> K {+0..7, +16..23},
// lanes 16-31 -> K {+8..15, +24..31}; p already includes the per-lane base.
__device__ __forceinline__ v16bf load_afrag(const __bf16* __restrict__ p) {
  v8bf lo = *(const v8bf*)p;
  v8bf hi = *(const v8bf*)(p + 16);
  v16bf a;
#pragma unroll
  for (int i = 0; i < 8; ++i) { a[i] = lo[i]; a[i + 8] = hi[i]; }
  return a;
}

// ---------------- kernels ----------------

__global__ void cvt_bf16_kernel(const float* __restrict__ s, __bf16* __restrict__ d, long n) {
  long i = (long)blockIdx.x * blockDim.x + threadIdx.x;
  if (i < n) d[i] = f2bf(s[i]);
}

__global__ void zero_bf_kernel(__bf16* __restrict__ d, long n) {
  long i = (long)blockIdx.x * blockDim.x + threadIdx.x;
  if (i < n) d[i] = f2bf(0.0f);
}

__global__ void zero_h_kernel(float* __restrict__ hf, __bf16* __restrict__ hb, int n) {
  int i = blockIdx.x * blockDim.x + threadIdx.x;
  if (i < n) { hf[i] = 0.0f; hb[i] = f2bf(0.0f); }
}

// C[M,N] = A[M,K] @ W[N,K]^T + bias[N], bf16 in, f32 accumulate.
// Each wave computes a 64x32 macro-tile: 4 M-frags x 2 N-frags = 8 WMMAs per
// k-step, reusing A-frags across both B-frags and B-frags across all 4 A-frags
// (24 B/lane of L2 traffic per WMMA vs 64 without blocking).
// Requires M % 64 == 0. W has Npad (mult of 32) rows; stores guarded by col<N.
__global__ void gemm_bf16_wmma(const __bf16* __restrict__ A, const __bf16* __restrict__ W,
                               const float* __restrict__ bias, float* __restrict__ C,
                               int M, int N, int Npad, int Kd, long ldc, int tiles) {
  int wave = threadIdx.x >> 5;
  int lane = threadIdx.x & 31;
  int gw = blockIdx.x * 8 + wave;
  if (gw >= tiles) return;                  // wave-uniform: EXEC all-ones for WMMA
  int nblks = Npad >> 5;
  int tm = (gw / nblks) << 6;
  int tn = (gw % nblks) << 5;
  int l16 = lane & 15;
  bool hi = lane >= 16;

  const __bf16* ar[4];
#pragma unroll
  for (int r = 0; r < 4; ++r) ar[r] = A + (long)(tm + 16 * r + l16) * Kd;
  const __bf16* wr0 = W + (long)(tn + l16) * Kd;
  const __bf16* wr1 = wr0 + (long)16 * Kd;

  v8f acc[4][2];
#pragma unroll
  for (int r = 0; r < 4; ++r) { acc[r][0] = (v8f){}; acc[r][1] = (v8f){}; }

  for (int k0 = 0; k0 < Kd; k0 += 32) {
    int ka = k0 + (hi ? 8 : 0);
    int kb = k0 + (hi ? 16 : 0);
    // B fragments: same K order as A: lanes 0-15 -> K {+0..7, +16..23}; lanes 16-31 -> {+8..15, +24..31}
    v16bf b0 = load_afrag(wr0 + ka);
    v16bf b1 = load_afrag(wr1 + ka);
#pragma unroll
    for (int r = 0; r < 4; ++r) {
      v16bf a = load_afrag(ar[r] + ka);
      acc[r][0] = __builtin_amdgcn_wmma_f32_16x16x32_bf16(
          false, a, false, b0, (short)0, acc[r][0], false, false);
      acc[r][1] = __builtin_amdgcn_wmma_f32_16x16x32_bf16(
          false, a, false, b1, (short)0, acc[r][1], false, false);
    }
  }

  // C/D layout: lane n (0-15) VGPR r -> (tileM+r, tileN+n); lane n+16 -> (+8, n)
  int col0 = tn + l16;
  int col1 = col0 + 16;
  bool ok0 = col0 < N, ok1 = col1 < N;
  float bv0 = ok0 ? bias[col0] : 0.0f;
  float bv1 = ok1 ? bias[col1] : 0.0f;
#pragma unroll
  for (int r = 0; r < 4; ++r) {
    int rb = tm + 16 * r + (hi ? 8 : 0);
#pragma unroll
    for (int i = 0; i < 8; ++i) {
      if (ok0) C[(long)(rb + i) * ldc + col0] = acc[r][0][i] + bv0;
      if (ok1) C[(long)(rb + i) * ldc + col1] = acc[r][1][i] + bv1;
    }
  }
}

// Additive attention: scores -> softmax -> weighted sum of features -> bf16 ctx.
// One block (256 threads, 8 waves) per batch element.
__global__ void attention_kernel(const float* __restrict__ q, const float* __restrict__ keys,
                                 const float* __restrict__ Va, const float* __restrict__ bva,
                                 const float* __restrict__ feat, __bf16* __restrict__ ctx_bf) {
  __shared__ float s_q[Hc];
  __shared__ float s_sc[Lc];
  int b = blockIdx.x, tid = threadIdx.x;
  int wave = tid >> 5, lane = tid & 31;

  for (int i = tid; i < Hc; i += 256) s_q[i] = q[(long)b * Hc + i];
  __syncthreads();

  // scores[l] = Va . tanh(q + keys[b,l]) + bva  (one wave per l, shuffle-reduce)
  for (int l = wave; l < Lc; l += 8) {
    const float* kr = keys + ((long)b * Lc + l) * Hc;
    float acc = 0.0f;
    for (int i = lane; i < Hc; i += 32) acc += Va[i] * tanhf(s_q[i] + kr[i]);
#pragma unroll
    for (int off = 16; off > 0; off >>= 1) acc += __shfl_xor(acc, off, 32);
    if (lane == 0) s_sc[l] = acc + bva[0];
  }
  __syncthreads();

  // softmax (redundant per thread; L=49 is tiny, read-only -> no race)
  float m = -1e30f;
  for (int l = 0; l < Lc; ++l) m = fmaxf(m, s_sc[l]);
  float ssum = 0.0f;
  for (int l = 0; l < Lc; ++l) ssum += __expf(s_sc[l] - m);
  float inv = 1.0f / ssum;

  const float* fb = feat + (long)b * Lc * Kc;
  for (int k = tid; k < Kc; k += 256) {
    float acc = 0.0f;
    for (int l = 0; l < Lc; ++l) acc += __expf(s_sc[l] - m) * fb[(long)l * Kc + k];
    ctx_bf[(long)b * Kc + k] = f2bf(acc * inv);
  }
}

// x = concat(emb[captions[:,t]], ctx_proj) in bf16. One block per batch element.
__global__ void build_x_kernel(const __bf16* __restrict__ emb_bf, const int* __restrict__ captions,
                               int t, const float* __restrict__ ctxp, __bf16* __restrict__ x_bf) {
  int b = blockIdx.x, tid = threadIdx.x;
  int tok = captions[b * Tc + t];
  const __bf16* e = emb_bf + (long)tok * Ec;
  __bf16* xr = x_bf + (long)b * (Ec + Hc);
  for (int i = tid; i < Ec; i += 256) {
    xr[i] = e[i];
    xr[Ec + i] = f2bf(ctxp[(long)b * Hc + i]);
  }
}

// GRU gate fusion: r,z,n + state update; emits fp32 h_new and bf16 copy.
__global__ void gru_gates_kernel(const float* __restrict__ gi, const float* __restrict__ gh,
                                 const float* __restrict__ h, float* __restrict__ h_new,
                                 __bf16* __restrict__ h_new_bf) {
  int idx = blockIdx.x * blockDim.x + threadIdx.x;
  if (idx >= Bc * Hc) return;
  int b = idx >> 9, j = idx & (Hc - 1);
  const float* gib = gi + (long)b * 3 * Hc;
  const float* ghb = gh + (long)b * 3 * Hc;
  float r = 1.0f / (1.0f + __expf(-(gib[j] + ghb[j])));
  float z = 1.0f / (1.0f + __expf(-(gib[Hc + j] + ghb[Hc + j])));
  float n = tanhf(gib[2 * Hc + j] + r * ghb[2 * Hc + j]);
  float hv = h[idx];
  float hn = (1.0f - z) * n + z * hv;
  h_new[idx] = hn;
  h_new_bf[idx] = f2bf(hn);
}

// ---------------- host launcher ----------------

extern "C" void kernel_launch(void* const* d_in, const int* in_sizes, int n_in,
                              void* d_out, int out_size, void* d_ws, size_t ws_size,
                              hipStream_t stream) {
  (void)in_sizes; (void)n_in; (void)out_size; (void)ws_size;

  const float* features = (const float*)d_in[0];
  const int*   captions = (const int*)d_in[1];
  const float* emb = (const float*)d_in[2];
  const float* Wq  = (const float*)d_in[3];
  const float* bq  = (const float*)d_in[4];
  const float* Uk  = (const float*)d_in[5];
  const float* bk  = (const float*)d_in[6];
  const float* Va  = (const float*)d_in[7];
  const float* bva = (const float*)d_in[8];
  const float* W0  = (const float*)d_in[9];
  const float* b0  = (const float*)d_in[10];
  const float* Wih = (const float*)d_in[11];
  const float* bih = (const float*)d_in[12];
  const float* Whh = (const float*)d_in[13];
  const float* bhh = (const float*)d_in[14];
  const float* W1  = (const float*)d_in[15];
  const float* b1  = (const float*)d_in[16];
  float* out = (float*)d_out;

  char* ws = (char*)d_ws;
  size_t off = 0;
  auto carve = [&](size_t bytes) -> char* {
    char* p = ws + off;
    off += (bytes + 255) & ~(size_t)255;
    return p;
  };

  __bf16* Wq_bf   = (__bf16*)carve((size_t)Hc * Hc * 2);
  __bf16* Uk_bf   = (__bf16*)carve((size_t)Hc * Kc * 2);
  __bf16* W0_bf   = (__bf16*)carve((size_t)Hc * Kc * 2);
  __bf16* Wih_bf  = (__bf16*)carve((size_t)3 * Hc * (Ec + Hc) * 2);
  __bf16* Whh_bf  = (__bf16*)carve((size_t)3 * Hc * Hc * 2);
  __bf16* W1_bf   = (__bf16*)carve((size_t)VcPad * Hc * 2);   // padded to 10016 rows
  __bf16* emb_bf  = (__bf16*)carve((size_t)Vc * Ec * 2);
  __bf16* feat_bf = (__bf16*)carve((size_t)Bc * Lc * Kc * 2);
  float*  keys    = (float*)carve((size_t)Bc * Lc * Hc * 4);
  float*  hA      = (float*)carve((size_t)Bc * Hc * 4);
  float*  hB      = (float*)carve((size_t)Bc * Hc * 4);
  __bf16* hbfA    = (__bf16*)carve((size_t)Bc * Hc * 2);
  __bf16* hbfB    = (__bf16*)carve((size_t)Bc * Hc * 2);
  float*  qbuf    = (float*)carve((size_t)Bc * Hc * 4);
  __bf16* ctx_bf  = (__bf16*)carve((size_t)Bc * Kc * 2);
  float*  ctxp    = (float*)carve((size_t)Bc * Hc * 4);
  __bf16* xbf     = (__bf16*)carve((size_t)Bc * (Ec + Hc) * 2);
  float*  gi      = (float*)carve((size_t)Bc * 3 * Hc * 4);
  float*  gh      = (float*)carve((size_t)Bc * 3 * Hc * 4);

  auto cvt = [&](const float* s, __bf16* d, long n) {
    cvt_bf16_kernel<<<(unsigned)((n + 255) / 256), 256, 0, stream>>>(s, d, n);
  };
  // C = A @ W^T + bias; M % 64 == 0; W buffer must have Npad = ceil32(N) rows.
  auto gemm = [&](const __bf16* A, const __bf16* Wt, const float* bias, float* C,
                  int M, int N, int Kd, long ldc) {
    int Npad = (N + 31) & ~31;
    int tiles = (M >> 6) * (Npad >> 5);
    gemm_bf16_wmma<<<(tiles + 7) / 8, 256, 0, stream>>>(A, Wt, bias, C, M, N, Npad, Kd, ldc, tiles);
  };

  // One-time preprocessing (L2-resident thereafter)
  cvt(Wq,  Wq_bf,  (long)Hc * Hc);
  cvt(Uk,  Uk_bf,  (long)Hc * Kc);
  cvt(W0,  W0_bf,  (long)Hc * Kc);
  cvt(Wih, Wih_bf, (long)3 * Hc * (Ec + Hc));
  cvt(Whh, Whh_bf, (long)3 * Hc * Hc);
  cvt(W1,  W1_bf,  (long)Vc * Hc);
  // zero the 16 pad rows of W1_bf so the padded N-tile contributes nothing
  zero_bf_kernel<<<((VcPad - Vc) * Hc + 255) / 256, 256, 0, stream>>>(
      W1_bf + (long)Vc * Hc, (long)(VcPad - Vc) * Hc);
  cvt(emb, emb_bf, (long)Vc * Ec);
  cvt(features, feat_bf, (long)Bc * Lc * Kc);

  zero_h_kernel<<<(Bc * Hc + 255) / 256, 256, 0, stream>>>(hA, hbfA, Bc * Hc);

  // keys_proj[B*L, H] = features @ Uk^T + bk   (M = 12544 = 64*196)
  gemm(feat_bf, Uk_bf, bk, keys, Bc * Lc, Hc, Kc, Hc);

  float*  h_f = hA;  __bf16* h_b = hbfA;
  float* hn_f = hB;  __bf16* hn_b = hbfB;

  for (int t = 0; t < Tc; ++t) {
    // q = h @ Wq^T + bq
    gemm(h_b, Wq_bf, bq, qbuf, Bc, Hc, Hc, Hc);
    // attention -> ctx (bf16)
    attention_kernel<<<Bc, 256, 0, stream>>>(qbuf, keys, Va, bva, features, ctx_bf);
    // ctx_proj = ctx @ W0^T + b0
    gemm(ctx_bf, W0_bf, b0, ctxp, Bc, Hc, Kc, Hc);
    // x = [emb_t ; ctx_proj]
    build_x_kernel<<<Bc, 256, 0, stream>>>(emb_bf, captions, t, ctxp, xbf);
    // gi = x @ Wih^T + bih ; gh = h @ Whh^T + bhh
    gemm(xbf, Wih_bf, bih, gi, Bc, 3 * Hc, Ec + Hc, 3 * Hc);
    gemm(h_b, Whh_bf, bhh, gh, Bc, 3 * Hc, Hc, 3 * Hc);
    // GRU update
    gru_gates_kernel<<<(Bc * Hc + 255) / 256, 256, 0, stream>>>(gi, gh, h_f, hn_f, hn_b);
    // logits[:, t, :] = h_new @ W1^T + b1  (64x32 macro-tiles over 256x10016)
    gemm(hn_b, W1_bf, b1, out + (long)t * Vc, Bc, Vc, Hc, (long)Tc * Vc);

    // ping-pong h
    float*  tf = h_f; h_f = hn_f; hn_f = tf;
    __bf16* tb = h_b; h_b = hn_b; hn_b = tb;
  }
}
